// ExphormerAttention_75651553951848
// MI455X (gfx1250) — compile-verified, matrix operand order checked by reading, not run on hardware
//
#include <hip/hip_runtime.h>

#define N_NODES 50000
#define N_EDGES 1600000
#define DIM_H   128
#define N_HEADS 8
// HEAD_D = 16, one WMMA N-tile per head
#define NODE_TILES (N_NODES / 16)   // 3125 exactly
#define EDGE_TILES (N_EDGES / 16)   // 100000 exactly

typedef __attribute__((ext_vector_type(16))) __bf16        v16bf;
typedef __attribute__((ext_vector_type(8)))  float         v8f;
typedef __attribute__((ext_vector_type(4)))  float         v4f;
typedef __attribute__((ext_vector_type(4)))  unsigned int  v4u;
typedef __attribute__((ext_vector_type(4)))  unsigned int  u32x4;
typedef __attribute__((ext_vector_type(8)))  int           i32x8;
typedef __attribute__((ext_vector_type(4)))  int           i32x4;

union BF16Frag {
  v16bf v;
  unsigned short u[16];
  v4u q[2];
};

// fp32 -> bf16 round-to-nearest-even
__device__ __forceinline__ unsigned short f2bf(float f) {
  unsigned int u = __float_as_uint(f);
  u += 0x7FFFu + ((u >> 16) & 1u);
  return (unsigned short)(u >> 16);
}

// Build one 16x32 bf16 A fragment for this lane from a row-major fp32 row.
// 16-bit A layout: lanes 0-15 hold row m=lane, K in {0..7, 16..23};
// lanes 16-31 hold row m=lane-16, K in {8..15, 24..31}.
__device__ __forceinline__ v16bf load_a_tile(const float* __restrict__ row,
                                             int kt, int half) {
  const v4f* p0 = (const v4f*)(row + kt * 32 + half * 8);
  const v4f* p1 = (const v4f*)(row + kt * 32 + 16 + half * 8);
  v4f a0 = p0[0], a1 = p0[1];
  v4f b0 = p1[0], b1 = p1[1];
  BF16Frag f;
  f.u[0]  = f2bf(a0.x); f.u[1]  = f2bf(a0.y); f.u[2]  = f2bf(a0.z); f.u[3]  = f2bf(a0.w);
  f.u[4]  = f2bf(a1.x); f.u[5]  = f2bf(a1.y); f.u[6]  = f2bf(a1.z); f.u[7]  = f2bf(a1.w);
  f.u[8]  = f2bf(b0.x); f.u[9]  = f2bf(b0.y); f.u[10] = f2bf(b0.z); f.u[11] = f2bf(b0.w);
  f.u[12] = f2bf(b1.x); f.u[13] = f2bf(b1.y); f.u[14] = f2bf(b1.z); f.u[15] = f2bf(b1.w);
  return f.v;
}

// ---------------------------------------------------------------------------
// Kernel 1: convert 4 fp32 128x128 weights into bf16 B-fragment layout.
// Fragment (kt, nt): 32 lanes x 16 bf16 each, lane holds column n = nt*16 +
// (lane&15); K distribution mirrors the A layout per half-wave.
// ---------------------------------------------------------------------------
__global__ void prep_weights(const float* __restrict__ WQ, const float* __restrict__ WK,
                             const float* __restrict__ WV, const float* __restrict__ WE,
                             unsigned short* __restrict__ out) {
  int tid = blockIdx.x * blockDim.x + threadIdx.x;
  if (tid >= 4 * 16384) return;
  int w    = tid >> 14;
  int r    = tid & 16383;
  int j    = r & 15;
  int lane = (r >> 4) & 31;
  int ntkt = r >> 9;          // (kt*8 + nt), 0..31
  int nt   = ntkt & 7;
  int kt   = ntkt >> 3;
  int half = lane >> 4;
  int nn   = lane & 15;
  int k = kt * 32 + ((j < 8) ? (half * 8 + j) : (16 + half * 8 + (j - 8)));
  int n = nt * 16 + nn;
  const float* W = (w == 0) ? WQ : (w == 1) ? WK : (w == 2) ? WV : WE;
  out[tid] = f2bf(W[k * DIM_H + n]);
}

// ---------------------------------------------------------------------------
// Kernel 2: node projections Qh/Kh/Vh = h @ {WQ,WK,WV}, bf16 WMMA, fp32 accum.
// One wave per 16-row tile; 3 * 8 * 4 = 96 v_wmma per wave.
// ---------------------------------------------------------------------------
__global__ void __launch_bounds__(256) node_proj(const float* __restrict__ h,
                                                 const unsigned short* __restrict__ Wb,
                                                 float* __restrict__ Qh,
                                                 float* __restrict__ Kh,
                                                 float* __restrict__ Vh) {
  int lane = threadIdx.x & 31;
  int wave = threadIdx.x >> 5;
  int tile = blockIdx.x * 8 + wave;
  if (tile >= NODE_TILES) return;
  int half = lane >> 4;
  int nn   = lane & 15;

  const float* hrow = h + (tile * 16 + nn) * DIM_H;
  v16bf A[4];
#pragma unroll
  for (int kt = 0; kt < 4; ++kt) A[kt] = load_a_tile(hrow, kt, half);

  float* outs[3] = {Qh, Kh, Vh};
#pragma unroll
  for (int w = 0; w < 3; ++w) {
    const unsigned short* WB = Wb + w * 16384;
    float* out = outs[w];
    for (int nt = 0; nt < 8; ++nt) {
      v8f acc = {0.f, 0.f, 0.f, 0.f, 0.f, 0.f, 0.f, 0.f};
#pragma unroll
      for (int kt = 0; kt < 4; ++kt) {
        BF16Frag bf;
        const v4u* bp = (const v4u*)(WB + ((kt * 8 + nt) * 32 + lane) * 16);
        bf.q[0] = bp[0]; bf.q[1] = bp[1];
        acc = __builtin_amdgcn_wmma_f32_16x16x32_bf16(
            false, A[kt], false, bf.v, (short)0, acc, false, false);
      }
      int col = nt * 16 + nn;
#pragma unroll
      for (int g = 0; g < 8; ++g) {
        int m = half * 8 + g;   // C layout: vgpr g, half selects row block
        out[(tile * 16 + m) * DIM_H + col] = acc[g];
      }
    }
  }
}

// ---------------------------------------------------------------------------
// Kernel 3: zero the atomic accumulators (wV then Z, contiguous)
// ---------------------------------------------------------------------------
__global__ void zero_ws(float* __restrict__ p, int n) {
  int i = blockIdx.x * blockDim.x + threadIdx.x;
  if (i < n) p[i] = 0.f;
}

// ---------------------------------------------------------------------------
// Kernel 4: fused edge projection + attention score + scatter.
// Per wave: 16 edges. The 32KB bf16 WE fragment blob is DMA'd into LDS by the
// Tensor Data Mover (one descriptor, issued by wave 0, completion tracked via
// TENSORcnt). Eh tile computed via WMMA; each 16-col accumulator tile == one
// head. Score via 16-lane shuffle reduction, scatter with
// global_atomic_add_f32 (lands in L2; wV fits).
// ---------------------------------------------------------------------------
__global__ void __launch_bounds__(256) edge_attn(
    const float* __restrict__ edge_attr,
    const unsigned short* __restrict__ WEb,
    const float* __restrict__ Qh, const float* __restrict__ Kh,
    const float* __restrict__ Vh,
    const long long* __restrict__ src_idx, const long long* __restrict__ dst_idx,
    float* __restrict__ wV, float* __restrict__ Z) {
  __shared__ v4u sWE4[2048];           // 32KB bf16 WE in fragment layout

#if __has_builtin(__builtin_amdgcn_tensor_load_to_lds)
  if ((threadIdx.x >> 5) == 0) {
    // D# describing the 32KB blob as one row of 8192 x 4B elements.
    unsigned lds_base = (unsigned)(unsigned long long)(__SIZE_TYPE__)&sWE4[0];
    unsigned long long ga = (unsigned long long)(__SIZE_TYPE__)WEb;
    u32x4 g0;
    g0.x = 1u;                                    // count=1 (valid user D#)
    g0.y = lds_base;                              // lds_addr (bytes)
    g0.z = (unsigned)(ga & 0xFFFFFFFFu);          // global_addr[31:0]
    g0.w = (unsigned)((ga >> 32) & 0x01FFFFFFu)   // global_addr[56:32]
           | (2u << 30);                          // type=2 ("image")
    i32x8 g1;
    g1[0] = 0x00020000;   // workgroup_mask=0, data_size=2 (4B), flags=0
    g1[1] = 0x20000000;   // tensor_dim0=0x2000 (low 16 bits at [63:48])
    g1[2] = 0x00010000;   // tensor_dim1=1 (bits [95:80])
    g1[3] = 0x20000000;   // tile_dim0=0x2000 (bits [127:112])
    g1[4] = 0x00000001;   // tile_dim1=1, tile_dim2=0
    g1[5] = 0x00002000;   // tensor_dim0_stride=8192 (low 32 of [207:160])
    g1[6] = 0x20000000;   // tensor_dim1_stride low bits at [223:208]
    g1[7] = 0;
    i32x4 gz = {0, 0, 0, 0};
#if defined(__clang_major__) && __clang_major__ >= 23
    i32x8 gz8 = {0, 0, 0, 0, 0, 0, 0, 0};
    __builtin_amdgcn_tensor_load_to_lds(g0, g1, gz, gz, gz8, 0);
#else
    __builtin_amdgcn_tensor_load_to_lds(g0, g1, gz, gz, 0);
#endif
#if __has_builtin(__builtin_amdgcn_s_wait_tensorcnt)
    __builtin_amdgcn_s_wait_tensorcnt(0);
#endif
  }
#else
  for (int i = threadIdx.x; i < 2048; i += 256)
    sWE4[i] = ((const v4u*)WEb)[i];
#endif
  __syncthreads();
  const unsigned short* sWE = (const unsigned short*)sWE4;

  int lane = threadIdx.x & 31;
  int wave = threadIdx.x >> 5;
  int tile = blockIdx.x * 8 + wave;    // grid sized exactly: tile < EDGE_TILES
  int e0   = tile * 16;
  int half = lane >> 4;
  int nn   = lane & 15;

  int my_src = (int)src_idx[e0 + nn];
  int my_dst = (int)dst_idx[e0 + nn];

  const float* erow = edge_attr + (e0 + nn) * DIM_H;
  if (e0 + 16 < N_EDGES)               // speculative prefetch of next tile
    __builtin_prefetch(edge_attr + (e0 + 16 + nn) * DIM_H, 0, 0);

  v16bf A[4];
#pragma unroll
  for (int kt = 0; kt < 4; ++kt) A[kt] = load_a_tile(erow, kt, half);

  const float inv_sqrt_d = 0.25f;      // 1/sqrt(16)
  for (int nt = 0; nt < 8; ++nt) {     // nt == head index
    v8f eacc = {0.f, 0.f, 0.f, 0.f, 0.f, 0.f, 0.f, 0.f};
#pragma unroll
    for (int kt = 0; kt < 4; ++kt) {
      BF16Frag bf;
      const v4u* bp = (const v4u*)(sWE + ((kt * 8 + nt) * 32 + lane) * 16);
      bf.q[0] = bp[0]; bf.q[1] = bp[1];
      eacc = __builtin_amdgcn_wmma_f32_16x16x32_bf16(
          false, A[kt], false, bf.v, (short)0, eacc, false, false);
    }
    int col = nt * 16 + nn;
#pragma unroll
    for (int g = 0; g < 8; ++g) {
      int m  = half * 8 + g;                 // edge row held in eacc[g]
      int sm = __shfl(my_src, m, 16);
      int dm = __shfl(my_dst, m, 16);
      float p = Kh[sm * DIM_H + col] * Qh[dm * DIM_H + col] * eacc[g];
      // sum over the 16 head dims (one per lane in this half-wave)
      for (int off = 8; off > 0; off >>= 1)
        p += __shfl_xor(p, off, 16);
      float s = __expf(fminf(fmaxf(p * inv_sqrt_d, -5.f), 5.f));
      float v = Vh[sm * DIM_H + col];
      atomicAdd(&wV[dm * DIM_H + col], v * s);
      if (nn == 0) atomicAdd(&Z[dm * N_HEADS + nt], s);
    }
  }
}

// ---------------------------------------------------------------------------
// Kernel 5: h_out = wV / (Z + 1e-6)
// ---------------------------------------------------------------------------
__global__ void normalize_out(const float* __restrict__ wV,
                              const float* __restrict__ Z,
                              float* __restrict__ out) {
  int i = blockIdx.x * blockDim.x + threadIdx.x;
  if (i < N_NODES * DIM_H) {
    int n  = i >> 7;
    int hd = (i & 127) >> 4;
    out[i] = wV[i] / (Z[n * N_HEADS + hd] + 1e-6f);
  }
}

extern "C" void kernel_launch(void* const* d_in, const int* in_sizes, int n_in,
                              void* d_out, int out_size, void* d_ws, size_t ws_size,
                              hipStream_t stream) {
  const float* h  = (const float*)d_in[0];
  const float* ea = (const float*)d_in[1];
  const float* WQ = (const float*)d_in[2];
  const float* WK = (const float*)d_in[3];
  const float* WV = (const float*)d_in[4];
  const float* WE = (const float*)d_in[5];
  const long long* eidx = (const long long*)d_in[6];   // int64 per reference
  const long long* src = eidx;
  const long long* dst = eidx + N_EDGES;

  // Workspace layout (~104.5 MB):
  //   [0, 128KB)       bf16 weights (WQ,WK,WV,WE fragment layout)
  //   Qh, Kh, Vh       3 x 25.6 MB fp32
  //   wV (25.6MB) + Z (1.6MB), contiguous for one zero pass
  char* ws = (char*)d_ws;
  unsigned short* Wb = (unsigned short*)ws;
  float* Qh  = (float*)(ws + (128 << 10));
  float* Kh  = Qh  + (size_t)N_NODES * DIM_H;
  float* Vh  = Kh  + (size_t)N_NODES * DIM_H;
  float* wVb = Vh  + (size_t)N_NODES * DIM_H;
  float* Zb  = wVb + (size_t)N_NODES * DIM_H;

  prep_weights<<<256, 256, 0, stream>>>(WQ, WK, WV, WE, Wb);
  node_proj<<<(NODE_TILES + 7) / 8, 256, 0, stream>>>(h, Wb, Qh, Kh, Vh);
  int zn = N_NODES * DIM_H + N_NODES * N_HEADS;
  zero_ws<<<(zn + 255) / 256, 256, 0, stream>>>(wVb, zn);
  edge_attn<<<EDGE_TILES / 8, 256, 0, stream>>>(ea, Wb + 3 * 16384, Qh, Kh, Vh,
                                                src, dst, wVb, Zb);
  normalize_out<<<(N_NODES * DIM_H + 255) / 256, 256, 0, stream>>>(wVb, Zb,
                                                                   (float*)d_out);
}
